// Diff_prop_18425409699925
// MI455X (gfx1250) — matrix operand order repaired and checked by reading, not compiled, not run on hardware
//
#include <hip/hip_runtime.h>
#include <hip/hip_bf16.h>

#define C_DIM  256
#define N_ROWS 1024
#define BATCH  8
#define F_EPS  1e-12f
#define LN_EPS 1e-5f
#define LEAK   0.01f

typedef __attribute__((ext_vector_type(16))) _Float16 v16h;
typedef __attribute__((ext_vector_type(8)))  _Float16 v8h;
typedef __attribute__((ext_vector_type(4)))  _Float16 v4h;
typedef __attribute__((ext_vector_type(8)))  float    v8f;

union FragU { v16h v; struct { v8h lo, hi; } p; };

// A-fragment (16x32 f16, MxK): lane = row (lane&15); lanes<16 hold K in
// [0,8)+[16,24), lanes>=16 hold K in [8,16)+[24,32). Two b128 loads.
__device__ __forceinline__ v16h load_a_frag(const _Float16* rowp, int k0, int lane) {
    const int kb = (lane & 16) ? 8 : 0;
    FragU f;
    f.p.lo = *(const v8h*)(rowp + k0 + kb);
    f.p.hi = *(const v8h*)(rowp + k0 + kb + 16);
    return f.v;
}

// B-fragment (32x16 f16, KxN): lane = col (lane&15); lanes<16 hold K in
// [0,8)+[16,24), lanes>=16 hold K in [8,16)+[24,32). Two b128 loads.
__device__ __forceinline__ v16h load_b_frag(const _Float16* colp, int k0, int lane) {
    const int kb = (lane & 16) ? 8 : 0;
    FragU f;
    f.p.lo = *(const v8h*)(colp + k0 + kb);
    f.p.hi = *(const v8h*)(colp + k0 + kb + 16);
    return f.v;
}

__device__ __forceinline__ v8f wmma_f16(v16h a, v16h b, v8f c) {
    return __builtin_amdgcn_wmma_f32_16x16x32_f16(false, a, false, b, (short)0, c, false, false);
}

// ---------------------------------------------------------------------------
// Stage A: per-row L2 normalize; emit xn (f16), xn*diag(W_adj) (f16), and
// x transposed per batch xT[c][row] (f16) for the Ax GEMM B-operand.
// One wave per row, 8 waves per block.
__global__ void prep_norm_kernel(const float* __restrict__ x,
                                 const float* __restrict__ Wadj,
                                 _Float16* __restrict__ xn,
                                 _Float16* __restrict__ xnd,
                                 _Float16* __restrict__ xT) {
    const int wid  = threadIdx.x >> 5;
    const int lane = threadIdx.x & 31;
    const int row  = blockIdx.x * 8 + wid;      // 0..8191
    const int b    = row >> 10;
    const int i    = row & 1023;
    const float* xr = x + (size_t)row * C_DIM;
    const int c0 = lane * 8;

    float4 a0 = *(const float4*)(xr + c0);
    float4 a1 = *(const float4*)(xr + c0 + 4);
    float v[8] = {a0.x, a0.y, a0.z, a0.w, a1.x, a1.y, a1.z, a1.w};

    float ss = 0.f;
    #pragma unroll
    for (int j = 0; j < 8; ++j) ss += v[j] * v[j];
    #pragma unroll
    for (int o = 16; o; o >>= 1) ss += __shfl_xor(ss, o);
    const float inv = 1.0f / fmaxf(sqrtf(ss), F_EPS);

    v8h hn, hd;
    #pragma unroll
    for (int j = 0; j < 8; ++j) {
        float d  = Wadj[(size_t)(c0 + j) * (C_DIM + 1)];   // diagonal of W_adj
        float nv = v[j] * inv;
        hn[j] = (_Float16)nv;
        hd[j] = (_Float16)(nv * d);
    }
    *(v8h*)(xn  + (size_t)row * C_DIM + c0) = hn;
    *(v8h*)(xnd + (size_t)row * C_DIM + c0) = hd;
    #pragma unroll
    for (int j = 0; j < 8; ++j)
        xT[((size_t)(b * C_DIM + c0 + j)) * N_ROWS + i] = (_Float16)v[j];
}

__global__ void wconv_kernel(const float* __restrict__ W, _Float16* __restrict__ Wh) {
    const int t = blockIdx.x * blockDim.x + threadIdx.x;   // 16384 threads, 4 each
    float4 w = ((const float4*)W)[t];
    v4h h = {(_Float16)w.x, (_Float16)w.y, (_Float16)w.z, (_Float16)w.w};
    *(v4h*)(Wh + (size_t)t * 4) = h;
}

// ---------------------------------------------------------------------------
// Stage B: S = (xn*diag) @ xn^T, per batch 1024x1024, K=256. One wave per
// 16x16 tile; 8 waves in a block share the same M-tile rows (L0/L2 reuse).
__global__ void gram_kernel(const _Float16* __restrict__ xnd,
                            const _Float16* __restrict__ xn,
                            float* __restrict__ S) {
    const int lane = threadIdx.x & 31;
    const int wid  = threadIdx.x >> 5;
    const int b  = blockIdx.z;
    const int m0 = blockIdx.y * 16;
    const int n0 = (blockIdx.x * 8 + wid) * 16;
    const _Float16* arow = xnd + ((size_t)(b * N_ROWS + m0 + (lane & 15))) * C_DIM;
    const _Float16* brow = xn  + ((size_t)(b * N_ROWS + n0 + (lane & 15))) * C_DIM;

    v8f acc = {};
    #pragma unroll
    for (int k0 = 0; k0 < C_DIM; k0 += 32)
        acc = wmma_f16(load_a_frag(arow, k0, lane), load_b_frag(brow, k0, lane), acc);

    const int nn = n0 + (lane & 15);
    const int mb = m0 + ((lane >> 4) << 3);
    float* srow = S + ((size_t)(b * N_ROWS + mb)) * N_ROWS + nn;
    #pragma unroll
    for (int r = 0; r < 8; ++r) srow[(size_t)r * N_ROWS] = acc[r];
}

// ---------------------------------------------------------------------------
// Stage C: per row: max, exp(5(s-max)), zero diagonal, L1 normalize.
// A stored as f16 (halves the dominant traffic); rowsum kept f32.
__global__ void rownorm_kernel(const float* __restrict__ S,
                               _Float16* __restrict__ A,
                               float* __restrict__ rowsum) {
    const int row = blockIdx.x;          // 0..8191
    const int i   = row & 1023;
    const int t    = threadIdx.x;        // 256 threads, 4 elements each
    const int lane = t & 31, wid = t >> 5;
    const float* srow = S + (size_t)row * N_ROWS;

    float4 v = ((const float4*)srow)[t];
    float e[4] = {v.x, v.y, v.z, v.w};

    __shared__ float red[8];
    __shared__ float bc[1];

    float m = fmaxf(fmaxf(e[0], e[1]), fmaxf(e[2], e[3]));
    #pragma unroll
    for (int o = 16; o; o >>= 1) m = fmaxf(m, __shfl_xor(m, o));
    if (lane == 0) red[wid] = m;
    __syncthreads();
    if (t == 0) {
        float mm = red[0];
        #pragma unroll
        for (int j = 1; j < 8; ++j) mm = fmaxf(mm, red[j]);
        bc[0] = mm;
    }
    __syncthreads();
    const float M = bc[0];
    __syncthreads();

    const int c0 = t * 4;
    float s = 0.f;
    #pragma unroll
    for (int j = 0; j < 4; ++j) {
        e[j] = __expf(5.0f * (e[j] - M));
        if (c0 + j == i) e[j] = 0.0f;    // zero diagonal
        s += e[j];
    }
    #pragma unroll
    for (int o = 16; o; o >>= 1) s += __shfl_xor(s, o);
    if (lane == 0) red[wid] = s;
    __syncthreads();
    if (t == 0) {
        float tt = 0.f;
        #pragma unroll
        for (int j = 0; j < 8; ++j) tt += red[j];
        float denom = fmaxf(tt, F_EPS);
        rowsum[row] = tt / denom;        // == 1 unless degenerate
        bc[0] = 1.0f / denom;
    }
    __syncthreads();
    const float inv = bc[0];

    v4h h;
    #pragma unroll
    for (int j = 0; j < 4; ++j) h[j] = (_Float16)(e[j] * inv);
    *(v4h*)(A + (size_t)row * N_ROWS + c0) = h;
}

// ---------------------------------------------------------------------------
// Stage D: Ax = A @ x (M=1024, K=1024, N=256) via WMMA; epilogue fuses
// u = rowsum*x - Ax (the collapsed form of the reference's diff pre-GEMM).
__global__ void ax_kernel(const _Float16* __restrict__ A,
                          const _Float16* __restrict__ xT,
                          const float* __restrict__ x,
                          const float* __restrict__ rowsum,
                          _Float16* __restrict__ u) {
    const int lane = threadIdx.x & 31;
    const int wid  = threadIdx.x >> 5;
    const int b  = blockIdx.z;
    const int m0 = blockIdx.y * 16;
    const int n0 = (blockIdx.x * 8 + wid) * 16;     // channel tile
    const _Float16* arow = A  + ((size_t)(b * N_ROWS + m0 + (lane & 15))) * N_ROWS;
    const _Float16* brow = xT + ((size_t)(b * C_DIM  + n0 + (lane & 15))) * N_ROWS;

    v8f acc = {};
    for (int k0 = 0; k0 < N_ROWS; k0 += 32)
        acc = wmma_f16(load_a_frag(arow, k0, lane), load_b_frag(brow, k0, lane), acc);

    const int nn = n0 + (lane & 15);
    const int mb = m0 + ((lane >> 4) << 3);
    #pragma unroll
    for (int r = 0; r < 8; ++r) {
        const int m = mb + r;
        const float rs = rowsum[b * N_ROWS + m];
        const float uv = rs * x[((size_t)(b * N_ROWS + m)) * C_DIM + nn] - acc[r];
        u[((size_t)(b * N_ROWS + m)) * C_DIM + nn] = (_Float16)uv;
    }
}

// ---------------------------------------------------------------------------
// Stage F: diff = u @ W^T + rowsum*b, then fused LayerNorm + LeakyReLU.
// 16 waves cover a full 16-row x 256-col block so LN fits in the epilogue.
__global__ void out_kernel(const _Float16* __restrict__ u,
                           const _Float16* __restrict__ Wh,
                           const float* __restrict__ rowsum,
                           const float* __restrict__ b_aff,
                           const float* __restrict__ gamma,
                           const float* __restrict__ beta,
                           float* __restrict__ out) {
    __shared__ float tile[16][264];
    const int lane = threadIdx.x & 31;
    const int wid  = threadIdx.x >> 5;          // 0..15
    const int b  = blockIdx.y;
    const int m0 = blockIdx.x * 16;
    const int n0 = wid * 16;
    const _Float16* arow = u  + ((size_t)(b * N_ROWS + m0 + (lane & 15))) * C_DIM;
    const _Float16* brow = Wh + (size_t)(n0 + (lane & 15)) * C_DIM;  // B[k][n]=W[n][k]

    v8f acc = {};
    #pragma unroll
    for (int k0 = 0; k0 < C_DIM; k0 += 32)
        acc = wmma_f16(load_a_frag(arow, k0, lane), load_b_frag(brow, k0, lane), acc);

    const int nn  = n0 + (lane & 15);
    const int mbl = (lane >> 4) << 3;
    const float bv = b_aff[nn];
    #pragma unroll
    for (int r = 0; r < 8; ++r)
        tile[mbl + r][nn] = acc[r] + rowsum[b * N_ROWS + m0 + mbl + r] * bv;
    __syncthreads();

    // wave `wid` owns output row m0+wid
    const int m = m0 + wid;
    const float* trow = tile[wid];
    float vals[8], s = 0.f, ss = 0.f;
    #pragma unroll
    for (int j = 0; j < 8; ++j) {
        float vv = trow[lane * 8 + j];
        vals[j] = vv; s += vv; ss += vv * vv;
    }
    #pragma unroll
    for (int o = 16; o; o >>= 1) { s += __shfl_xor(s, o); ss += __shfl_xor(ss, o); }
    const float mu   = s * (1.0f / C_DIM);
    const float var  = ss * (1.0f / C_DIM) - mu * mu;
    const float rsig = rsqrtf(var + LN_EPS);
    #pragma unroll
    for (int j = 0; j < 8; ++j) {
        const int c = lane * 8 + j;
        float y = (vals[j] - mu) * rsig * gamma[c] + beta[c];
        y = (y > 0.f) ? y : LEAK * y;
        out[((size_t)(b * N_ROWS + m)) * C_DIM + c] = y;
    }
}

// ---------------------------------------------------------------------------
extern "C" void kernel_launch(void* const* d_in, const int* in_sizes, int n_in,
                              void* d_out, int out_size, void* d_ws, size_t ws_size,
                              hipStream_t stream) {
    const float* x     = (const float*)d_in[0];   // local_feat [8,16,64,256]
    // d_in[1] global_feat, d_in[2] pos : unused by the reference
    const float* Wadj  = (const float*)d_in[3];   // [256,256] (only diagonal used)
    const float* Waff  = (const float*)d_in[4];   // [256,256]
    const float* baff  = (const float*)d_in[5];   // [256]
    const float* gamma = (const float*)d_in[6];   // [256]
    const float* beta  = (const float*)d_in[7];   // [256]
    float* out = (float*)d_out;

    char* ws = (char*)d_ws;
    const size_t SZ_ROWC_H = (size_t)BATCH * N_ROWS * C_DIM * sizeof(_Float16); // 4 MB
    _Float16* xn     = (_Float16*)(ws);                         // 4 MB
    _Float16* xnd    = (_Float16*)(ws + SZ_ROWC_H);             // 4 MB
    _Float16* xT     = (_Float16*)(ws + 2 * SZ_ROWC_H);         // 4 MB
    _Float16* u      = (_Float16*)(ws + 3 * SZ_ROWC_H);         // 4 MB
    _Float16* Wh     = (_Float16*)(ws + 4 * SZ_ROWC_H);         // 128 KB
    float*    rowsum = (float*)   (ws + 4 * SZ_ROWC_H + (size_t)C_DIM * C_DIM * 2); // 32 KB
    char*     p      = ws + 4 * SZ_ROWC_H + (size_t)C_DIM * C_DIM * 2
                          + (size_t)BATCH * N_ROWS * sizeof(float);
    float*    S      = (float*)p;                               // 32 MB
    _Float16* A      = (_Float16*)(p + (size_t)BATCH * N_ROWS * N_ROWS * sizeof(float)); // 16 MB

    prep_norm_kernel<<<dim3(BATCH * N_ROWS / 8), dim3(256), 0, stream>>>(x, Wadj, xn, xnd, xT);
    wconv_kernel<<<dim3(64), dim3(256), 0, stream>>>(Waff, Wh);
    gram_kernel<<<dim3(8, 64, BATCH), dim3(256), 0, stream>>>(xnd, xn, S);
    rownorm_kernel<<<dim3(BATCH * N_ROWS), dim3(256), 0, stream>>>(S, A, rowsum);
    ax_kernel<<<dim3(2, 64, BATCH), dim3(256), 0, stream>>>(A, xT, x, rowsum, u);
    out_kernel<<<dim3(64, BATCH), dim3(512), 0, stream>>>(u, Wh, rowsum, baff, gamma, beta, out);
}
